// RegionFeatureLearner_15719580303601
// MI455X (gfx1250) — compile-verified, matrix operand order checked by reading, not run on hardware
//
#include <hip/hip_runtime.h>
#include <hip/hip_bf16.h>
#include <stddef.h>

// ---------------------------------------------------------------------------
// RegionFeatureLearner for MI455X (gfx1250, wave32, WMMA).
//
// Per block: one region r, one 128-sample tile. The recurrent GEMM
// g = [x_t ; h_{t-1}] @ [Wih | Whh]^T is done with v_wmma_f32_16x16x32_f16,
// A (activations, f16) and B (weights, f16) both resident in LDS.
// Each wave owns one 16-row M-tile end-to-end, so the time loop uses only
// wave-local ordering (s_wait_dscnt) — no block barriers after init.
// Activations use raw v_rcp_f32 / v_tanh_f32 (no IEEE-div fixup chain).
// ---------------------------------------------------------------------------

typedef __attribute__((ext_vector_type(16))) _Float16 v16h;
typedef __attribute__((ext_vector_type(8)))  _Float16 v8h;
typedef __attribute__((ext_vector_type(8)))  float    v8f;

#define REGIONS 16
#define HIDDEN  100
#define INPUT   5
#define SEQLEN  62
#define G4      400          // 4*HIDDEN gate rows
#define MTILE   128          // samples per block
#define KPAD    128          // padded K: x at [0,5), h at [8,108)
#define NSAMP   (128*256)    // 32768
#define TMAX    6

// region index table (flattened) + offsets
__constant__ int REG_OFF[REGIONS + 1] =
    {0, 5, 10, 12, 14, 20, 26, 31, 36, 41, 43, 45, 50, 53, 56, 59, 62};
__constant__ int REG_IDX[SEQLEN] = {
    3, 0, 1, 2, 4,
    7, 8, 9, 10, 11,
    5, 6,
    13, 12,
    14, 15, 23, 24, 32, 33,
    22, 21, 31, 30, 40, 39,
    16, 17, 18, 19, 20,
    25, 26, 27, 28, 29,
    34, 35, 36, 37, 38,
    41, 42,
    49, 48,
    43, 44, 45, 46, 47,
    50, 51, 57,
    56, 55, 61,
    52, 53, 54,
    58, 59, 60};

// ---- LDS layout (bytes) ----------------------------------------------------
#define OFF_W     0                                   // f16 [400][128]  102400
#define OFF_A     (OFF_W  + G4 * KPAD * 2)            // f16 [128][128]   32768
#define OFF_HS    (OFF_A  + MTILE * KPAD * 2)         // f32 [128][100]   51200
#define OFF_CS    (OFF_HS + MTILE * HIDDEN * 4)       // f32 [128][100]   51200
#define OFF_XR    (OFF_CS + MTILE * HIDDEN * 4)       // f32 [128][6][5]  15360
#define OFF_BF    (OFF_XR + MTILE * TMAX * INPUT * 4) // f32 [400]         1600
#define OFF_WB    (OFF_BF + G4 * 4)                   // f32 [400][5]      8000
#define OFF_BB    (OFF_WB + G4 * INPUT * 4)           // f32 [400]         1600
#define OFF_SCR   (OFF_BB + G4 * 4)                   // f32 [8][256]      8192
#define SMEM_BYTES (OFF_SCR + 8 * 256 * 4)            // = 272320

// Fast transcendentals: raw hardware v_rcp_f32 / v_exp_f32 / v_tanh_f32,
// no IEEE-division fixup sequences.
__device__ __forceinline__ float fast_rcp(float x) {
    return __builtin_amdgcn_rcpf(x);
}
__device__ __forceinline__ float sigm(float x) {
    return fast_rcp(1.0f + __expf(-x));
}
#if __has_builtin(__builtin_amdgcn_tanhf)
__device__ __forceinline__ float tanh_fast(float x) {
    return __builtin_amdgcn_tanhf(x);            // v_tanh_f32 (gfx1250)
}
#else
__device__ __forceinline__ float tanh_fast(float x) {
    float e = __expf(2.0f * x);
    return (e - 1.0f) * fast_rcp(e + 1.0f);
}
#endif

// Load one 16x32 f16 fragment (A or B operand) from a row-major [*, KPAD]
// LDS matrix, per the CDNA5 16-bit operand layout:
//   lanes 0-15  : VGPR0-3 -> K=kb+0..7,  VGPR4-7 -> K=kb+16..23
//   lanes 16-31 : VGPR0-3 -> K=kb+8..15, VGPR4-7 -> K=kb+24..31
// => two ds_load_b128 at (kb + c0) and (kb + 16 + c0), c0 = 8*(lane>=16).
__device__ __forceinline__ v16h load_frag(const _Float16* base, int row, int kb, int c0) {
    const _Float16* p = base + row * KPAD + kb + c0;
    v8h lo = *(const v8h*)(p);
    v8h hi = *(const v8h*)(p + 16);
    return __builtin_shufflevector(lo, hi, 0, 1, 2, 3, 4, 5, 6, 7,
                                           8, 9, 10, 11, 12, 13, 14, 15);
}

__global__ void region_lstm_kernel(const float* __restrict__ features,
                                   const float* __restrict__ w_ih,
                                   const float* __restrict__ w_hh,
                                   const float* __restrict__ b_ih,
                                   const float* __restrict__ b_hh,
                                   float* __restrict__ out) {
    extern __shared__ char smem[];
    _Float16* W    = (_Float16*)(smem + OFF_W);   // interleaved [4u+gate][k]
    _Float16* A    = (_Float16*)(smem + OFF_A);   // [m][k] activations
    float*    Hs   = (float*)(smem + OFF_HS);
    float*    Cs   = (float*)(smem + OFF_CS);
    float*    xr   = (float*)(smem + OFF_XR);     // [m][6][5]
    float*    bfw  = (float*)(smem + OFF_BF);
    float*    Wb   = (float*)(smem + OFF_WB);
    float*    bbw  = (float*)(smem + OFF_BB);
    float*    scrb = (float*)(smem + OFF_SCR);

    const int tid = threadIdx.x;
    const int r   = blockIdx.y;
    const int m0  = blockIdx.x * MTILE;
    const int off = REG_OFF[r];
    const int T   = REG_OFF[r + 1] - off;

    // ---------------- init: weights / biases / x-slice / state --------------
    // Forward weights, f16, gate-interleaved rows: p = 4*u + gate,
    // source row j = gate*100 + u.  k<5 -> Wih, 8<=k<108 -> Whh, else 0.
    for (int e = tid; e < G4 * KPAD; e += 256) {
        int p = e >> 7, k = e & (KPAD - 1);
        int u = p >> 2, gate = p & 3;
        int j = gate * HIDDEN + u;
        float v = 0.0f;
        if (k < INPUT)
            v = w_ih[((size_t)(r * 2) * G4 + j) * INPUT + k];
        else if (k >= 8 && k < 8 + HIDDEN)
            v = w_hh[((size_t)(r * 2) * G4 + j) * HIDDEN + (k - 8)];
        W[e] = (_Float16)v;
    }
    for (int e = tid; e < MTILE * KPAD; e += 256) A[e] = (_Float16)0.0f;
    for (int e = tid; e < MTILE * HIDDEN; e += 256) { Hs[e] = 0.0f; Cs[e] = 0.0f; }
    // x slice for this region/tile: xr[m][t][k]
    for (int e = tid; e < MTILE * T * INPUT; e += 256) {
        int m = e / (T * INPUT);
        int t = (e / INPUT) % T;
        int k = e % INPUT;
        xr[(m * TMAX + t) * INPUT + k] =
            features[((size_t)(m0 + m) * SEQLEN + REG_IDX[off + t]) * INPUT + k];
    }
    // forward bias (interleaved order), backward weights/bias (natural order)
    for (int e = tid; e < G4; e += 256) {
        int u = e >> 2, gate = e & 3;
        int j = gate * HIDDEN + u;
        bfw[e] = b_ih[(size_t)(r * 2) * G4 + j] + b_hh[(size_t)(r * 2) * G4 + j];
        bbw[e] = b_ih[(size_t)(r * 2 + 1) * G4 + e] + b_hh[(size_t)(r * 2 + 1) * G4 + e];
    }
    for (int e = tid; e < G4 * INPUT; e += 256)
        Wb[e] = w_ih[(size_t)(r * 2 + 1) * G4 * INPUT + e];
    __syncthreads();

    // ---------------- recurrent loop: each wave owns M-tile w ---------------
    const int w    = tid >> 5;
    const int lane = tid & 31;
    const int lm   = lane & 15;
    const int hi16 = lane >> 4;
    const int c0   = hi16 * 8;
    const int mrow = w * 16 + lm;          // this lane's A row
    float* scr = scrb + w * 256;           // per-wave 16x16 f32 scratch

    for (int t = 0; t < T; ++t) {
        // write x_t into A rows of this wave's M-tile (k = 0..4)
        {
            const float* xp = &xr[(mrow * TMAX + t) * INPUT];
            _Float16* ap = &A[mrow * KPAD];
            if (hi16 == 0) {
                ap[0] = (_Float16)xp[0];
                ap[1] = (_Float16)xp[1];
                ap[2] = (_Float16)xp[2];
            } else {
                ap[3] = (_Float16)xp[3];
                ap[4] = (_Float16)xp[4];
            }
        }
        __builtin_amdgcn_wave_barrier();
        asm volatile("s_wait_dscnt 0x0" ::: "memory");

        // A fragments for this M-tile, all 4 K-chunks (reused over 25 N-tiles)
        v16h af[4];
#pragma unroll
        for (int kc = 0; kc < 4; ++kc) af[kc] = load_frag(A, mrow, kc * 32, c0);

        for (int n = 0; n < 25; ++n) {
            v8f acc = {};
#pragma unroll
            for (int kc = 0; kc < 4; ++kc) {
                v16h bf = load_frag(W, n * 16 + lm, kc * 32, c0);
                acc = __builtin_amdgcn_wmma_f32_16x16x32_f16(
                    false, af[kc], false, bf, (short)0, acc, false, false);
            }
            // spill 16x16 tile (+bias) to wave scratch in (row, col) order
            float bcol = bfw[n * 16 + lm];
#pragma unroll
            for (int v = 0; v < 8; ++v)
                scr[(v + 8 * hi16) * 16 + lm] = acc[v] + bcol;
            __builtin_amdgcn_wave_barrier();
            asm volatile("s_wait_dscnt 0x0" ::: "memory");

            // LSTM cell update for the 4 hidden units covered by this N-tile
#pragma unroll
            for (int rep = 0; rep < 2; ++rep) {
                int idx = lane * 2 + rep;        // 0..63
                int ms  = idx & 15;
                int uo  = idx >> 4;              // 0..3
                int row = w * 16 + ms;
                int u   = n * 4 + uo;
                float iv = scr[ms * 16 + uo * 4 + 0];
                float fv = scr[ms * 16 + uo * 4 + 1];
                float gv = scr[ms * 16 + uo * 4 + 2];
                float ov = scr[ms * 16 + uo * 4 + 3];
                float cold = Cs[row * HIDDEN + u];
                float cn = sigm(fv) * cold + sigm(iv) * tanh_fast(gv);
                float hn = sigm(ov) * tanh_fast(cn);
                Cs[row * HIDDEN + u] = cn;
                Hs[row * HIDDEN + u] = hn;
                A[row * KPAD + 8 + u] = (_Float16)hn;   // feeds next step's GEMM
            }
            __builtin_amdgcn_wave_barrier();
            asm volatile("s_wait_dscnt 0x0" ::: "memory");
        }
    }
    __syncthreads();

    // ---------------- backward direction (1 step, h0=c0=0) + store ----------
    for (int e = tid; e < MTILE * HIDDEN; e += 256) {
        int m = e / HIDDEN, u = e % HIDDEN;
        const float* xl = &xr[(m * TMAX + (T - 1)) * INPUT];
        float g4[4];
#pragma unroll
        for (int gate = 0; gate < 4; ++gate) {
            int j = gate * HIDDEN + u;
            const float* wrow = &Wb[j * INPUT];
            float s = bbw[j];
#pragma unroll
            for (int k = 0; k < INPUT; ++k) s += xl[k] * wrow[k];
            g4[gate] = s;
        }
        float cn = sigm(g4[0]) * tanh_fast(g4[2]);   // f-gate * c0 == 0
        float hb = sigm(g4[3]) * tanh_fast(cn);
        float* op = out + ((size_t)(m0 + m) * REGIONS + r) * (2 * HIDDEN);
        op[u]          = Hs[m * HIDDEN + u];
        op[HIDDEN + u] = hb;
    }
}

extern "C" void kernel_launch(void* const* d_in, const int* in_sizes, int n_in,
                              void* d_out, int out_size, void* d_ws, size_t ws_size,
                              hipStream_t stream) {
    (void)in_sizes; (void)n_in; (void)d_ws; (void)ws_size; (void)out_size;
    const float* features = (const float*)d_in[0];
    const float* w_ih     = (const float*)d_in[1];
    const float* w_hh     = (const float*)d_in[2];
    const float* b_ih     = (const float*)d_in[3];
    const float* b_hh     = (const float*)d_in[4];
    float* out = (float*)d_out;

    dim3 grid(NSAMP / MTILE, REGIONS);   // 256 x 16 blocks
    dim3 block(256);                     // 8 wave32s
    region_lstm_kernel<<<grid, block, SMEM_BYTES, stream>>>(
        features, w_ih, w_hh, b_ih, b_hh, out);
}
